// SimpleGATLayer_30262339568070
// MI455X (gfx1250) — matrix operand order repaired and checked by reading, not compiled, not run on hardware
//
#include <hip/hip_runtime.h>

typedef __attribute__((ext_vector_type(16))) _Float16 v16h;
typedef __attribute__((ext_vector_type(8)))  float    v8f;

#define KDIM 128
#define HD   64     // HEADS * OUT_DIM

// ---------- helpers ----------

// Monotonic float -> uint mapping so unsigned atomicMax == float max.
__device__ __forceinline__ unsigned fkey(float f) {
  unsigned b = __float_as_uint(f);
  return (b & 0x80000000u) ? ~b : (b | 0x80000000u);
}
__device__ __forceinline__ float funkey(unsigned k) {
  return __uint_as_float((k & 0x80000000u) ? (k ^ 0x80000000u) : ~k);
}
__device__ __forceinline__ float lrelu(float v) { return v > 0.f ? v : 0.01f * v; }

// A-fragment (16x32 f16, MxK): lane = M%16 row; lanes>=16 shift K by 8 within
// each 16-block. Element e -> k = (e>=8?16:0) + hoff + (e&7). Two contiguous
// 8-float runs -> vectorizable loads. p points at row start + kbase.
__device__ __forceinline__ v16h load_a_frag(const float* __restrict__ p, int hoff) {
  v16h r;
#pragma unroll
  for (int i = 0; i < 8; ++i) {
    r[i]     = (_Float16)p[hoff + i];
    r[8 + i] = (_Float16)p[16 + hoff + i];
  }
  return r;
}

// B-fragment (32x16 f16, KxN): lane = N col; same K order as A-fragment:
// element e -> k = (e>=8?16:0) + koff + (e&7), koff = 8 for lanes 16-31.
// p points at column's source row start + kbase (B[k][n] = W[n][k]).
__device__ __forceinline__ v16h load_b_frag(const float* __restrict__ p, int koff) {
  v16h r;
#pragma unroll
  for (int i = 0; i < 8; ++i) {
    r[i]     = (_Float16)p[koff + i];
    r[8 + i] = (_Float16)p[16 + koff + i];
  }
  return r;
}

// ---------- kernels ----------

__global__ void gat_init(float* __restrict__ out, unsigned* __restrict__ maxkey,
                         float* __restrict__ denom, int n64, int n4) {
  int i = blockIdx.x * blockDim.x + threadIdx.x;
  if (i < n64) out[i] = 0.f;
  if (i < n4) { maxkey[i] = 0u; denom[i] = 0.f; }  // key(most negative) == 0
}

// h = x @ W^T  via v_wmma_f32_16x16x32_f16.
// Block = 256 thr = 8 waves; wave -> 16 rows x 64 cols (4 WMMA tiles = 4 heads).
__global__ void __launch_bounds__(256)
gat_gemm_wmma(const float* __restrict__ x, const float* __restrict__ W,
              float* __restrict__ h, int N) {
  const int wave = threadIdx.x >> 5;
  const int lane = threadIdx.x & 31;
  const int lrow = lane & 15;
  const int hi   = (lane & 16) ? 1 : 0;
  const int hoff = hi ? 8 : 0;    // A-frag K sub-offset
  const int koff = hi ? 8 : 0;    // B-frag K sub-offset

  const int mBase = blockIdx.x * 128 + wave * 16;
  int arow = mBase + lrow;
  if (arow >= N) arow = N - 1;                  // clamp loads; stores guarded
  const float* xrow = x + (size_t)arow * KDIM;

  const float* w0 = W + (size_t)(0 * 16 + lrow) * KDIM;  // head-tile columns
  const float* w1 = W + (size_t)(1 * 16 + lrow) * KDIM;
  const float* w2 = W + (size_t)(2 * 16 + lrow) * KDIM;
  const float* w3 = W + (size_t)(3 * 16 + lrow) * KDIM;

  v8f acc0 = {}, acc1 = {}, acc2 = {}, acc3 = {};

#pragma unroll
  for (int ks = 0; ks < 4; ++ks) {
    const int kb = ks * 32;
    v16h a  = load_a_frag(xrow + kb, hoff);
    v16h b0 = load_b_frag(w0 + kb, koff);
    v16h b1 = load_b_frag(w1 + kb, koff);
    v16h b2 = load_b_frag(w2 + kb, koff);
    v16h b3 = load_b_frag(w3 + kb, koff);
    acc0 = __builtin_amdgcn_wmma_f32_16x16x32_f16(false, a, false, b0, (short)0, acc0, false, false);
    acc1 = __builtin_amdgcn_wmma_f32_16x16x32_f16(false, a, false, b1, (short)0, acc1, false, false);
    acc2 = __builtin_amdgcn_wmma_f32_16x16x32_f16(false, a, false, b2, (short)0, acc2, false, false);
    acc3 = __builtin_amdgcn_wmma_f32_16x16x32_f16(false, a, false, b3, (short)0, acc3, false, false);
  }

  // C/D layout: col = lane%16; VGPR r -> row r (lanes 0-15) / 8+r (lanes 16-31)
  const int rbase = mBase + (hi ? 8 : 0);
#pragma unroll
  for (int r = 0; r < 8; ++r) {
    const int orow = rbase + r;
    if (orow < N) {
      float* hp = h + (size_t)orow * HD;
      hp[ 0 + lrow] = acc0[r];
      hp[16 + lrow] = acc1[r];
      hp[32 + lrow] = acc2[r];
      hp[48 + lrow] = acc3[r];
    }
  }
}

// al[n,h] = h[n,h,:] . attn_l[h,:] ; ar likewise. One thread per (node, head).
__global__ void gat_node_logits(const float* __restrict__ h,
                                const float* __restrict__ attn_l,
                                const float* __restrict__ attn_r,
                                float* __restrict__ al, float* __restrict__ ar, int N) {
  int t = blockIdx.x * blockDim.x + threadIdx.x;
  if (t >= N * 4) return;
  const int n = t >> 2, hh = t & 3;
  const float* hp = h + (size_t)n * HD + hh * 16;
  const float* lp = attn_l + hh * 16;
  const float* rp = attn_r + hh * 16;
  float sl = 0.f, sr = 0.f;
#pragma unroll
  for (int d = 0; d < 16; ++d) { sl += hp[d] * lp[d]; sr += hp[d] * rp[d]; }
  al[t] = sl; ar[t] = sr;
}

// Pass 1: segment-max of leaky_relu(al[src]+ar[dst]) per (dst, head).
__global__ void gat_edge_max(const int* __restrict__ ei, const float* __restrict__ al,
                             const float* __restrict__ ar, unsigned* __restrict__ maxkey, int E) {
  int e = blockIdx.x * blockDim.x + threadIdx.x;
  if (e >= E) return;
  const int s = ei[e], d = ei[E + e];
  const float4 a = *(const float4*)(al + (size_t)s * 4);
  const float4 b = *(const float4*)(ar + (size_t)d * 4);
  atomicMax(&maxkey[d * 4 + 0], fkey(lrelu(a.x + b.x)));
  atomicMax(&maxkey[d * 4 + 1], fkey(lrelu(a.y + b.y)));
  atomicMax(&maxkey[d * 4 + 2], fkey(lrelu(a.z + b.z)));
  atomicMax(&maxkey[d * 4 + 3], fkey(lrelu(a.w + b.w)));
}

// Pass 2: denom[dst,h] += exp(score - m[dst,h])
__global__ void gat_edge_expsum(const int* __restrict__ ei, const float* __restrict__ al,
                                const float* __restrict__ ar, const unsigned* __restrict__ maxkey,
                                float* __restrict__ denom, int E) {
  int e = blockIdx.x * blockDim.x + threadIdx.x;
  if (e >= E) return;
  const int s = ei[e], d = ei[E + e];
  const float4 a = *(const float4*)(al + (size_t)s * 4);
  const float4 b = *(const float4*)(ar + (size_t)d * 4);
  float sc[4] = { lrelu(a.x + b.x), lrelu(a.y + b.y), lrelu(a.z + b.z), lrelu(a.w + b.w) };
#pragma unroll
  for (int hh = 0; hh < 4; ++hh) {
    const float m = funkey(maxkey[d * 4 + hh]);
    atomicAdd(&denom[d * 4 + hh], __expf(sc[hh] - m));
  }
}

// Pass 3: out[dst] += w[e,h] * h[src]  (64 f32 atomics / edge, L2-resident)
__global__ void gat_edge_scatter(const int* __restrict__ ei, const float* __restrict__ al,
                                 const float* __restrict__ ar, const unsigned* __restrict__ maxkey,
                                 const float* __restrict__ denom, const float* __restrict__ h,
                                 float* __restrict__ out, int E) {
  int e = blockIdx.x * blockDim.x + threadIdx.x;
  if (e >= E) return;
  const int s = ei[e], d = ei[E + e];
  const float4 a = *(const float4*)(al + (size_t)s * 4);
  const float4 b = *(const float4*)(ar + (size_t)d * 4);
  float sc[4] = { lrelu(a.x + b.x), lrelu(a.y + b.y), lrelu(a.z + b.z), lrelu(a.w + b.w) };
  float w[4];
#pragma unroll
  for (int hh = 0; hh < 4; ++hh) {
    const float m = funkey(maxkey[d * 4 + hh]);
    w[hh] = __expf(sc[hh] - m) / denom[d * 4 + hh];
  }
  const float* hs = h + (size_t)s * HD;
  float* od = out + (size_t)d * HD;
#pragma unroll
  for (int hh = 0; hh < 4; ++hh) {
#pragma unroll
    for (int q = 0; q < 4; ++q) {
      const float4 hv = *(const float4*)(hs + hh * 16 + q * 4);
      atomicAdd(&od[hh * 16 + q * 4 + 0], w[hh] * hv.x);
      atomicAdd(&od[hh * 16 + q * 4 + 1], w[hh] * hv.y);
      atomicAdd(&od[hh * 16 + q * 4 + 2], w[hh] * hv.z);
      atomicAdd(&od[hh * 16 + q * 4 + 3], w[hh] * hv.w);
    }
  }
}

__global__ void gat_elu(float* __restrict__ out, int n) {
  int i = blockIdx.x * blockDim.x + threadIdx.x;
  if (i >= n) return;
  const float v = out[i];
  out[i] = v > 0.f ? v : (__expf(v) - 1.f);
}

// ---------- launch ----------

extern "C" void kernel_launch(void* const* d_in, const int* in_sizes, int n_in,
                              void* d_out, int out_size, void* d_ws, size_t ws_size,
                              hipStream_t stream) {
  const float* x      = (const float*)d_in[0];
  const int*   ei     = (const int*)d_in[1];
  const float* W      = (const float*)d_in[2];
  const float* attn_l = (const float*)d_in[3];
  const float* attn_r = (const float*)d_in[4];
  float* out = (float*)d_out;

  const int N = in_sizes[0] / KDIM;   // 50000
  const int E = in_sizes[1] / 2;      // 800000

  // workspace layout (floats): h[N*64] | al[N*4] | ar[N*4] | maxkey[N*4] | denom[N*4]
  float*    h      = (float*)d_ws;
  float*    al     = h + (size_t)N * HD;
  float*    ar     = al + (size_t)N * 4;
  unsigned* maxkey = (unsigned*)(ar + (size_t)N * 4);
  float*    denom  = (float*)(maxkey + (size_t)N * 4);

  const int n64 = N * HD, n4 = N * 4;

  gat_init<<<(n64 + 255) / 256, 256, 0, stream>>>(out, maxkey, denom, n64, n4);
  gat_gemm_wmma<<<(N + 127) / 128, 256, 0, stream>>>(x, W, h, N);
  gat_node_logits<<<(n4 + 255) / 256, 256, 0, stream>>>(h, attn_l, attn_r, al, ar, N);
  gat_edge_max<<<(E + 255) / 256, 256, 0, stream>>>(ei, al, ar, maxkey, E);
  gat_edge_expsum<<<(E + 255) / 256, 256, 0, stream>>>(ei, al, ar, maxkey, denom, E);
  gat_edge_scatter<<<(E + 255) / 256, 256, 0, stream>>>(ei, al, ar, maxkey, denom, h, out, E);
  gat_elu<<<(n64 + 255) / 256, 256, 0, stream>>>(out, n64);
}
